// ContextualAttention_85779086836492
// MI455X (gfx1250) — matrix operand order repaired and checked by reading, not compiled, not run on hardware
//
#include <hip/hip_runtime.h>
#include <hip/hip_bf16.h>
#include <math.h>

// ---------------- problem constants (from reference) ----------------
#define HH   192
#define WW   192
#define CC   128
#define WS2  96
#define NN   9216           // number of patches (= n = m)
#define DD   512            // c * r * r
#define SOFTMAX_SCALE 10.0f

typedef __attribute__((ext_vector_type(16))) _Float16 v16h;
typedef __attribute__((ext_vector_type(8)))  _Float16 v8h;
typedef __attribute__((ext_vector_type(8)))  float    v8f;
typedef __attribute__((ext_vector_type(4)))  float    v4f;

__device__ __forceinline__ v16h cat16(v8h lo, v8h hi) {
  return __builtin_shufflevector(lo, hi, 0,1,2,3,4,5,6,7,8,9,10,11,12,13,14,15);
}

__device__ __forceinline__ v8f wmma_f16(v16h a, v16h b, v8f c) {
  // D = A(16x32 f16) * B(32x16 f16) + C(16x16 f32)
  return __builtin_amdgcn_wmma_f32_16x16x32_f16(false, a, false, b, (short)0, c, false, false);
}

// --------------------------------------------------------------------
// Prep: f16 Q (n x 128 row-major), f16 K=keys (m x 128 row-major),
// f16 V^T (512 x 9216), per-key validity. One block per patch m.
// --------------------------------------------------------------------
__global__ __launch_bounds__(128) void prep_kernel(
    const float* __restrict__ x, const float* __restrict__ mask,
    _Float16* __restrict__ q16, _Float16* __restrict__ k16,
    _Float16* __restrict__ vT, float* __restrict__ validb) {
  const int m = blockIdx.x;       // patch index
  const int c = threadIdx.x;      // channel
  const int i = m / WS2, j = m % WS2;

  const float* xc = x + ((size_t)c * HH + 2 * i) * WW + 2 * j;
  const float v00 = xc[0];
  const float v01 = xc[1];
  const float v10 = xc[WW];
  const float v11 = xc[WW + 1];
  const float f = v00;            // feat value (subsampled)

  float ss = f * f;
  ss += __shfl_xor(ss, 1, 32);
  ss += __shfl_xor(ss, 2, 32);
  ss += __shfl_xor(ss, 4, 32);
  ss += __shfl_xor(ss, 8, 32);
  ss += __shfl_xor(ss, 16, 32);
  __shared__ float red[4];
  if ((threadIdx.x & 31) == 0) red[threadIdx.x >> 5] = ss;
  __syncthreads();
  const float tot = red[0] + red[1] + red[2] + red[3];
  const float inv = 1.0f / (sqrtf(tot) + 1e-8f);

  q16[(size_t)m * CC + c] = (_Float16)f;
  k16[(size_t)m * CC + c] = (_Float16)(f * inv);

  const size_t vb = (size_t)(c * 4) * NN + m;      // d = c*4 + di*2 + dj
  vT[vb                 ] = (_Float16)v00;
  vT[vb + NN            ] = (_Float16)v01;
  vT[vb + 2 * (size_t)NN] = (_Float16)v10;
  vT[vb + 3 * (size_t)NN] = (_Float16)v11;

  if (threadIdx.x == 0)
    validb[m] = (mask[(size_t)(2 * i) * WW + 2 * j] == 0.0f) ? 1.0f : 0.0f;
}

// --------------------------------------------------------------------
// Fused flash-attention, cooperative scores:
//   block = 4 waves, one 16-row query tile, 64 keys per step.
//   wave w: S tile for keys [m0+16w, m0+16w+16)   (4 WMMAs)
//           local row max / row sum, masked P -> shared LDS
//   softmax state merged via log-sum-exp across waves (1 LDS exchange)
//   all waves: P(16x64) x V(64x128) into private 16x128 acc (16 WMMAs)
// WMMAs/wave/64 keys: 20 (vs 32 with redundant scores).
// --------------------------------------------------------------------
__global__ __launch_bounds__(128) void attn_kernel(
    const float* __restrict__ x, const float* __restrict__ mask,
    const _Float16* __restrict__ q16, const _Float16* __restrict__ k16,
    const _Float16* __restrict__ vT, const float* __restrict__ validb,
    float* __restrict__ out) {
  __shared__ _Float16 pbuf[16][64];   // shared masked P tile (row x key)
  __shared__ float    red[16][8];     // [row][0..3]=rowmax_w  [4..7]=rowsum_w

  const int lane = threadIdx.x & 31;
  const int wave = threadIdx.x >> 5;      // key sub-tile owner AND col group
  const int lh = lane >> 4;               // 0: lanes 0-15, 1: lanes 16-31
  const int ln = lane & 15;
  const int rowbase = blockIdx.x * 16;    // query-row tile base
  const int colbase = wave * 128;         // output (d) column base

  // ---- preload Q A-fragments (4 chunks of K=32 channels) -------------
  v16h aq[4];
  {
    const _Float16* qrow = q16 + (size_t)(rowbase + ln) * CC;
    #pragma unroll
    for (int kc4 = 0; kc4 < 4; ++kc4) {
      const int kc = kc4 * 32;
      v8h lo = *(const v8h*)(qrow + kc + lh * 8);
      v8h hi = *(const v8h*)(qrow + kc + 16 + lh * 8);
      aq[kc4] = cat16(lo, hi);
    }
  }

  v8f acc[8];
  #pragma unroll
  for (int t = 0; t < 8; ++t) { v8f z{}; acc[t] = z; }
  float mrun[8], lrun[8];
  #pragma unroll
  for (int g = 0; g < 8; ++g) { mrun[g] = -1e30f; lrun[g] = 0.0f; }

  for (int m0 = 0; m0 < NN; m0 += 64) {
    const int mykey = m0 + wave * 16;     // this wave's 16-key sub-tile

    // ---- S tile for own sub-tile: 4 WMMAs over 128 channels -------
    v8f s{};
    const _Float16* krow = k16 + (size_t)(mykey + ln) * CC;
    #pragma unroll
    for (int kc4 = 0; kc4 < 4; ++kc4) {
      const int kc = kc4 * 32;
      v8h lo = *(const v8h*)(krow + kc + lh * 8);
      v8h hi = *(const v8h*)(krow + kc + 16 + lh * 8);
      s = wmma_f16(aq[kc4], cat16(lo, hi), s);
    }
    const float valid = validb[mykey + ln];

    // ---- local (per-wave) row max and row sum of exp(a - rm) ------
    float rm[8], el[8], rs[8];
    #pragma unroll
    for (int g = 0; g < 8; ++g) {
      const float a = (valid != 0.0f) ? s[g] * SOFTMAX_SCALE : -10000.0f;
      float r = a;
      r = fmaxf(r, __shfl_xor(r, 1, 32));
      r = fmaxf(r, __shfl_xor(r, 2, 32));
      r = fmaxf(r, __shfl_xor(r, 4, 32));
      r = fmaxf(r, __shfl_xor(r, 8, 32));
      rm[g] = r;
      const float e = __expf(a - r);
      el[g] = e;
      float t = e;
      t += __shfl_xor(t, 1, 32);
      t += __shfl_xor(t, 2, 32);
      t += __shfl_xor(t, 4, 32);
      t += __shfl_xor(t, 8, 32);
      rs[g] = t;
    }
    // publish per-wave (rowmax, rowsum)
    if (ln == 0) {
      #pragma unroll
      for (int g = 0; g < 8; ++g) {
        const int row = g + lh * 8;
        red[row][wave]     = rm[g];
        red[row][4 + wave] = rs[g];
      }
    }
    __syncthreads();   // A: (rm, rs) visible

    // ---- merge softmax state across the 4 sub-tiles ----------------
    float pscale[8], pfac[8];
    #pragma unroll
    for (int g = 0; g < 8; ++g) {
      const int row = g + lh * 8;
      const v4f rmv = *(const v4f*)&red[row][0];
      const v4f rsv = *(const v4f*)&red[row][4];
      float nm = fmaxf(fmaxf(rmv[0], rmv[1]), fmaxf(rmv[2], rmv[3]));
      nm = fmaxf(nm, mrun[g]);
      const float ladd = rsv[0] * __expf(rmv[0] - nm)
                       + rsv[1] * __expf(rmv[1] - nm)
                       + rsv[2] * __expf(rmv[2] - nm)
                       + rsv[3] * __expf(rmv[3] - nm);
      const float sc = __expf(mrun[g] - nm);
      lrun[g] = lrun[g] * sc + ladd;     // denominator: UNmasked exp (ref)
      mrun[g] = nm;
      pscale[g] = sc;
      pfac[g] = __expf(rm[g] - nm) * valid;  // attn *= valid before P.V
    }

    // ---- write this wave's masked P into the shared tile ----------
    #pragma unroll
    for (int g = 0; g < 8; ++g) {
      const int row = g + lh * 8;
      pbuf[row][wave * 16 + ln] = (_Float16)(el[g] * pfac[g]);
    }
    __syncthreads();   // B: full P(16x64) visible

    // ---- rescale running accumulators ------------------------------
    #pragma unroll
    for (int t = 0; t < 8; ++t)
      #pragma unroll
      for (int g = 0; g < 8; ++g) acc[t][g] *= pscale[g];

    // ---- load shared P as two A-fragments (K = 64 keys) -----------
    const _Float16* pr = &pbuf[ln][0];
    v16h ap0, ap1;
    {
      v8h lo = *(const v8h*)(pr + lh * 8);
      v8h hi = *(const v8h*)(pr + 16 + lh * 8);
      ap0 = cat16(lo, hi);
      v8h lo2 = *(const v8h*)(pr + 32 + lh * 8);
      v8h hi2 = *(const v8h*)(pr + 48 + lh * 8);
      ap1 = cat16(lo2, hi2);
    }

    // ---- O += P x V : 8 column tiles x 2 key chunks ---------------
    #pragma unroll
    for (int t = 0; t < 8; ++t) {
      const _Float16* vrow = vT + (size_t)(colbase + t * 16 + ln) * NN + m0;
      v8h lo0 = *(const v8h*)(vrow + lh * 8);
      v8h hi0 = *(const v8h*)(vrow + 16 + lh * 8);
      acc[t] = wmma_f16(ap0, cat16(lo0, hi0), acc[t]);
      v8h lo1 = *(const v8h*)(vrow + 32 + lh * 8);
      v8h hi1 = *(const v8h*)(vrow + 48 + lh * 8);
      acc[t] = wmma_f16(ap1, cat16(lo1, hi1), acc[t]);
    }
    __syncthreads();   // C: protect pbuf/red from next iteration
  }

  // ---- epilogue: divide by softmax sum, scatter + blend -------------
  #pragma unroll
  for (int g = 0; g < 8; ++g) lrun[g] = 1.0f / lrun[g];

  #pragma unroll
  for (int t = 0; t < 8; ++t) {
    const int d = colbase + t * 16 + ln;
    const int c  = d >> 2;
    const int di = (d >> 1) & 1;
    const int dj = d & 1;
    #pragma unroll
    for (int g = 0; g < 8; ++g) {
      const int rrow = rowbase + g + lh * 8;     // patch index
      const int i = rrow / WS2, j = rrow % WS2;
      const int yy = 2 * i + di, xx = 2 * j + dj;
      const size_t oidx = ((size_t)c * HH + yy) * WW + xx;
      const float mk = mask[(size_t)yy * WW + xx];
      const float xv = x[oidx];
      const float rec = acc[t][g] * lrun[g];
      out[oidx] = xv * (1.0f - mk) + rec * mk;
    }
  }
}

// --------------------------------------------------------------------
extern "C" void kernel_launch(void* const* d_in, const int* in_sizes, int n_in,
                              void* d_out, int out_size, void* d_ws, size_t ws_size,
                              hipStream_t stream) {
  (void)in_sizes; (void)n_in; (void)out_size; (void)ws_size;
  const float* x    = (const float*)d_in[0];   // (1,128,192,192) f32
  const float* mask = (const float*)d_in[1];   // (1,1,192,192)  f32
  float* out = (float*)d_out;

  char* ws = (char*)d_ws;
  size_t off = 0;
  _Float16* q16 = (_Float16*)(ws + off); off += (size_t)NN * CC * 2;  // 2.25 MB
  _Float16* k16 = (_Float16*)(ws + off); off += (size_t)NN * CC * 2;  // 2.25 MB
  _Float16* vT  = (_Float16*)(ws + off); off += (size_t)DD * NN * 2;  // 9.0  MB
  float* validb = (float*)   (ws + off); off += (size_t)NN * 4;

  prep_kernel<<<NN, 128, 0, stream>>>(x, mask, q16, k16, vT, validb);
  attn_kernel<<<NN / 16, 128, 0, stream>>>(x, mask, q16, k16, vT, validb, out);
}
